// ClusteringLayer_84791244358066
// MI455X (gfx1250) — matrix operand order repaired and checked by reading, not compiled, not run on hardware
//
#include <hip/hip_runtime.h>

typedef __attribute__((ext_vector_type(16))) __bf16 v16bf;
typedef __attribute__((ext_vector_type(8)))  float  v8f;
typedef __attribute__((ext_vector_type(4)))  float  v4f;   // native vector: OK for nontemporal builtins

#define DIM    64
#define KCLUST 256
#define RSW    36   // LDS row stride in dwords (144 B; 36 mod 64 bank-stride -> conflict-free over 16 lanes)

union BFrag {
  v16bf v;
  unsigned u32[8];
};

// Truncating bf16 split of a pair of floats, packed into dwords with v_perm_b32.
// hi = {bf16_trunc(b), bf16_trunc(a)}, lo = {bf16_trunc(b - bhi), bf16_trunc(a - ahi)}.
__device__ __forceinline__ void split_pack_pair(float a, float b,
                                                unsigned& hi, unsigned& lo,
                                                float& ssq) {
  ssq = fmaf(a, a, ssq);
  ssq = fmaf(b, b, ssq);
  unsigned ba = __builtin_bit_cast(unsigned, a);
  unsigned bb = __builtin_bit_cast(unsigned, b);
  hi = __builtin_amdgcn_perm(bb, ba, 0x07060302u);  // [a.hi16 | b.hi16<<16]
  float ra = a - __builtin_bit_cast(float, ba & 0xFFFF0000u);
  float rb = b - __builtin_bit_cast(float, bb & 0xFFFF0000u);
  lo = __builtin_amdgcn_perm(__builtin_bit_cast(unsigned, rb),
                             __builtin_bit_cast(unsigned, ra), 0x07060302u);
}

// One block = 8 waves = 128 rows. One wave = 16 rows x all 256 clusters.
__global__ __launch_bounds__(256) void cluster_q_kernel(
    const float* __restrict__ x, const float* __restrict__ clusters,
    float* __restrict__ out) {
  __shared__ unsigned cl_hi[KCLUST * RSW];  // 36 KB, split-bf16 hi halves (2 elems/dword)
  __shared__ unsigned cl_lo[KCLUST * RSW];  // 36 KB, split-bf16 residuals
  __shared__ float    c_sq[KCLUST];

  const int tid = threadIdx.x;

  // ---- Phase 0: clusters -> split-bf16 LDS + exact fp32 c_sq (one row per thread) ----
  {
    const v4f* crow4 = (const v4f*)(clusters + tid * DIM);
    unsigned* hrow = cl_hi + tid * RSW;
    unsigned* lrow = cl_lo + tid * RSW;
    float s = 0.f;
#pragma unroll
    for (int j = 0; j < 16; ++j) {
      v4f f = crow4[j];
      unsigned h0, l0, h1, l1;
      split_pack_pair(f.x, f.y, h0, l0, s);
      split_pack_pair(f.z, f.w, h1, l1, s);
      *(uint2*)(hrow + j * 2) = uint2{h0, h1};
      *(uint2*)(lrow + j * 2) = uint2{l0, l1};
    }
    c_sq[tid] = s;
  }
  __syncthreads();

  const int wave = tid >> 5;
  const int lane = tid & 31;
  const int lr   = lane & 15;   // N within tile / A row selector
  const int hl   = lane >> 4;   // lane-half selector (A K-split & C M-split)

  const int  r0      = (blockIdx.x * 8 + wave) * 16;
  const long rowbase = (long)(r0 + lr) * DIM;

  // ---- A fragments (16x32 bf16, hi/lo split) + exact per-row ||x||^2 ----
  // 16-bit A layout: lane half hl holds K = {hl*8..+7, 16+hl*8..+7} of each 32-wide slice.
  BFrag ah[2], al[2];
  float ssq = 0.f;
#pragma unroll
  for (int kc = 0; kc < 2; ++kc) {
#pragma unroll
    for (int ch = 0; ch < 2; ++ch) {
      const int kbase = kc * 32 + ch * 16 + hl * 8;
      const v4f f0 = __builtin_nontemporal_load((const v4f*)(x + rowbase + kbase));
      const v4f f1 = __builtin_nontemporal_load((const v4f*)(x + rowbase + kbase + 4));
      split_pack_pair(f0.x, f0.y, ah[kc].u32[ch * 4 + 0], al[kc].u32[ch * 4 + 0], ssq);
      split_pack_pair(f0.z, f0.w, ah[kc].u32[ch * 4 + 1], al[kc].u32[ch * 4 + 1], ssq);
      split_pack_pair(f1.x, f1.y, ah[kc].u32[ch * 4 + 2], al[kc].u32[ch * 4 + 2], ssq);
      split_pack_pair(f1.z, f1.w, ah[kc].u32[ch * 4 + 3], al[kc].u32[ch * 4 + 3], ssq);
    }
  }
  ssq += __shfl_xor(ssq, 16, 32);  // combine halves -> full ||x_row||^2, row = r0+lr

  float base[8];  // 1 + ||x_m||^2 for the row handled by C-tile VGPR v in this half
#pragma unroll
  for (int v = 0; v < 8; ++v)
    base[v] = 1.0f + __shfl(ssq, v + 8 * hl, 32);

  // ---- main loop: 16 column tiles, all accumulators resident ----
  v8f   qt[16];
  float rsum[8] = {0.f, 0.f, 0.f, 0.f, 0.f, 0.f, 0.f, 0.f};
#pragma unroll
  for (int nt = 0; nt < 16; ++nt) {
    const int n = nt * 16 + lr;
    v8f acc = {};
#pragma unroll
    for (int kc = 0; kc < 2; ++kc) {
      // 16-bit B layout: lane half hl holds K = {hl*8..+7, 16+hl*8..+7} of the 32-slice, column = lr.
      BFrag bh, bl;
      const int off = n * RSW + kc * 16 + hl * 4;
      *(uint4*)(bh.u32 + 0) = *(const uint4*)(cl_hi + off);
      *(uint4*)(bh.u32 + 4) = *(const uint4*)(cl_hi + off + 8);
      *(uint4*)(bl.u32 + 0) = *(const uint4*)(cl_lo + off);
      *(uint4*)(bl.u32 + 4) = *(const uint4*)(cl_lo + off + 8);
      acc = __builtin_amdgcn_wmma_f32_16x16x32_bf16(false, ah[kc].v, false, bh.v,
                                                    (short)0, acc, false, false);
      acc = __builtin_amdgcn_wmma_f32_16x16x32_bf16(false, ah[kc].v, false, bl.v,
                                                    (short)0, acc, false, false);
      acc = __builtin_amdgcn_wmma_f32_16x16x32_bf16(false, al[kc].v, false, bh.v,
                                                    (short)0, acc, false, false);
    }
    const float csq = c_sq[n];
    v8f q = {};
#pragma unroll
    for (int v = 0; v < 8; ++v) {
      // 1 + ||x-c||^2 = (1 + x_sq + c_sq) - 2*cross ; ALPHA=1 -> q = 1/(1+d)
      float d  = fmaf(-2.0f, acc[v], base[v] + csq);
      float qq = __builtin_amdgcn_rcpf(d);
      q[v] = qq;
      rsum[v] += qq;
    }
    qt[nt] = q;
  }

  // ---- row-sum reduction across the 16 lanes of each half; rcp + 1 Newton step ----
#pragma unroll
  for (int v = 0; v < 8; ++v) {
    float s = rsum[v];
    s += __shfl_xor(s, 1, 32);
    s += __shfl_xor(s, 2, 32);
    s += __shfl_xor(s, 4, 32);
    s += __shfl_xor(s, 8, 32);
    float r = __builtin_amdgcn_rcpf(s);
    rsum[v] = r * fmaf(-s, r, 2.0f);
  }

  // ---- normalize + single streaming (non-temporal) store of q ----
#pragma unroll
  for (int nt = 0; nt < 16; ++nt) {
#pragma unroll
    for (int v = 0; v < 8; ++v) {
      __builtin_nontemporal_store(qt[nt][v] * rsum[v],
          out + (long)(r0 + v + 8 * hl) * KCLUST + nt * 16 + lr);
    }
  }
}

extern "C" void kernel_launch(void* const* d_in, const int* in_sizes, int n_in,
                              void* d_out, int out_size, void* d_ws, size_t ws_size,
                              hipStream_t stream) {
  (void)n_in; (void)out_size; (void)d_ws; (void)ws_size;
  const float* x        = (const float*)d_in[0];
  const float* clusters = (const float*)d_in[1];
  float* out            = (float*)d_out;

  const int n_rows = in_sizes[0] / DIM;      // 262144
  const int blocks = n_rows / 128;           // 2048 blocks, 8 waves each
  cluster_q_kernel<<<blocks, 256, 0, stream>>>(x, clusters, out);
}
